// Grid_TransformerBlock_36876589203661
// MI455X (gfx1250) — matrix operand order repaired and checked by reading, not compiled, not run on hardware
//
#include <hip/hip_runtime.h>
#include <hip/hip_bf16.h>

// ---------------- problem constants ----------------
constexpr int BB = 2;       // batch
constexpr int NN = 8192;    // points
constexpr int DP = 32;      // D_POINTS
constexpr int DM = 128;     // D_MODEL
constexpr int KK = 24;      // K neighbors
constexpr int PE = 60;      // PE_DIM
constexpr int LDA = 136;    // LDS A-matrix stride (bf16 elems), 16B aligned rows

typedef __attribute__((ext_vector_type(16))) __bf16 v16bf;
typedef __attribute__((ext_vector_type(8)))  __bf16 v8bf;
typedef __attribute__((ext_vector_type(8)))  float  v8f;

// -------- WMMA fragment loaders (CDNA5 16x16x32 bf16 layouts, ISA 7.12.2) ----
// A (16x32, MxK), row-major in LDS. lane<16: K = {0..7,16..23}; lane>=16: {8..15,24..31}
__device__ __forceinline__ v16bf load_a_lds(const __bf16* base, int lane, int k0) {
  int row = lane & 15;
  int kb  = k0 + ((lane >> 4) << 3);           // +0 or +8
  const __bf16* p = base + row * LDA + kb;
  v8bf lo = *(const v8bf*)(p);
  v8bf hi = *(const v8bf*)(p + 16);
  v16bf a;
#pragma unroll
  for (int i = 0; i < 8; ++i) { a[i] = lo[i]; a[8 + i] = hi[i]; }
  return a;
}

// B (32x16, KxN). W stored transposed [N][Kdim] row-major bf16; per lane K matches A:
// lane<16: K = {0..7,16..23}; lane>=16: {8..15,24..31}.
__device__ __forceinline__ v16bf load_b_glb(const __bf16* W, int ldw, int lane,
                                            int n0, int k0) {
  int n  = n0 + (lane & 15);
  int kb = k0 + ((lane >> 4) << 3);            // +0 or +8
  const v8bf* q = (const v8bf*)(W + n * ldw + kb);
  v8bf b0 = q[0], b1 = q[2];
  v16bf b;
#pragma unroll
  for (int i = 0; i < 8; ++i) { b[i] = b0[i]; b[8 + i] = b1[i]; }
  return b;
}

// 32xKdim (LDS, bf16) @ KdimxN-tile (global, bf16) -> two 16x16 f32 acc tiles
__device__ __forceinline__ void gemm_acc(const __bf16* Abuf, const __bf16* Wt,
                                         int Kdim, int lane, int n0,
                                         v8f& c0, v8f& c1) {
  v8f z = {};
  c0 = z; c1 = z;
  for (int ks = 0; ks < Kdim; ks += 32) {
    v16bf bf  = load_b_glb(Wt, Kdim, lane, n0, ks);
    v16bf af0 = load_a_lds(Abuf, lane, ks);
    v16bf af1 = load_a_lds(Abuf + 16 * LDA, lane, ks);
    c0 = __builtin_amdgcn_wmma_f32_16x16x32_bf16(false, af0, false, bf,
                                                 (short)0, c0, false, false);
    c1 = __builtin_amdgcn_wmma_f32_16x16x32_bf16(false, af1, false, bf,
                                                 (short)0, c1, false, false);
  }
}

// ---------------- kernel 0: weight convert (bf16, transposed, padded) --------
__global__ void wconv_kernel(const float* __restrict__ d1w,
                             const float* __restrict__ d2w,
                             const float* __restrict__ g1w,
                             const float* __restrict__ g2w,
                             __bf16* __restrict__ d1t, __bf16* __restrict__ d2t,
                             __bf16* __restrict__ g1t, __bf16* __restrict__ g2t) {
  const int total = DM * 64 + 3 * DM * DM;
  for (int i = blockIdx.x * blockDim.x + threadIdx.x; i < total;
       i += gridDim.x * blockDim.x) {
    if (i < DM * 64) {                       // d1: [60,128] -> [128][64] (pad K)
      int n = i >> 6, k = i & 63;
      d1t[i] = (__bf16)(k < PE ? d1w[k * DM + n] : 0.0f);
    } else {
      int e = i - DM * 64;
      int m = e >> 14;                       // which 128x128 matrix
      int r = e & 16383;
      int n = r >> 7, k = r & 127;
      const float* src = (m == 0) ? d2w : (m == 1) ? g1w : g2w;
      __bf16* dst      = (m == 0) ? d2t : (m == 1) ? g1t : g2t;
      dst[r] = (__bf16)src[k * DM + n];
    }
  }
}

// ---------------- kernel 1: x = features @ fc1_w + fc1_b ---------------------
__global__ __launch_bounds__(128) void xproj_kernel(
    const float* __restrict__ feats, const float* __restrict__ w,
    const float* __restrict__ bias, float* __restrict__ xbuf) {
  __shared__ float f[DP];
  const int p = blockIdx.x, d = threadIdx.x;
  if (d < DP) f[d] = feats[(size_t)p * DP + d];
  __syncthreads();
  float acc = bias[d];
#pragma unroll
  for (int j = 0; j < DP; ++j) acc += f[j] * w[j * DM + d];
  xbuf[(size_t)p * DM + d] = acc;
}

// ---------------- kernel 2: split brute-force KNN (top-24) -------------------
// 64 queries x 4 candidate-splits per block (256 threads). Each thread scans
// N/4 candidates into a register-resident sorted top-24; partial lists are
// merged 4-way in LDS. 256 blocks -> 2048 wave32s (vs 512 unsplit).
constexpr int QPB = 64;             // queries per block
constexpr int SPL = 4;              // candidate splits per query
constexpr int CHK = NN / SPL;       // candidates per split (2048)

__global__ __launch_bounds__(256) void knn_kernel(const float* __restrict__ xyz,
                                                  int* __restrict__ knn) {
  __shared__ float sx[SPL][QPB][3];
  __shared__ float pd[SPL][QPB][KK];
  __shared__ int   pi[SPL][QPB][KK];
  const int tid = threadIdx.x;
  const int ql  = tid & (QPB - 1);          // query within block
  const int s   = tid >> 6;                 // split id 0..3
  const int q   = blockIdx.x * QPB + ql;    // global query id
  const int b   = q / NN;
  const float qx = xyz[(size_t)q * 3 + 0];
  const float qy = xyz[(size_t)q * 3 + 1];
  const float qz = xyz[(size_t)q * 3 + 2];
  float bd[KK]; int bi[KK];
#pragma unroll
  for (int i = 0; i < KK; ++i) { bd[i] = __builtin_inff(); bi[i] = 0; }
  const size_t base = (size_t)b * NN * 3;
  const int c0 = s * CHK;
  for (int t0 = 0; t0 < CHK; t0 += QPB) {
    __syncthreads();
    {
      int cand = c0 + t0 + ql;
      sx[s][ql][0] = xyz[base + (size_t)cand * 3 + 0];
      sx[s][ql][1] = xyz[base + (size_t)cand * 3 + 1];
      sx[s][ql][2] = xyz[base + (size_t)cand * 3 + 2];
    }
    __syncthreads();
    for (int j = 0; j < QPB; ++j) {
      float dx = qx - sx[s][j][0], dy = qy - sx[s][j][1], dz = qz - sx[s][j][2];
      float d = dx * dx + dy * dy + dz * dz;
      if (d < bd[KK - 1]) {                 // register-resident insertion
        float cd = d; int ci = c0 + t0 + j;
#pragma unroll
        for (int i = 0; i < KK; ++i) {
          if (cd < bd[i]) {
            float td = bd[i]; int ti = bi[i];
            bd[i] = cd; bi[i] = ci; cd = td; ci = ti;
          }
        }
      }
    }
  }
#pragma unroll
  for (int i = 0; i < KK; ++i) { pd[s][ql][i] = bd[i]; pi[s][ql][i] = bi[i]; }
  __syncthreads();
  // 4-way stable merge (lower split wins ties -> index-ascending tie order)
  if (tid < QPB) {
    const int t = tid;
    int p0 = 0, p1 = 0, p2 = 0, p3 = 0;
    const float inf = __builtin_inff();
    size_t obase = (size_t)(blockIdx.x * QPB + t) * KK;
    for (int o = 0; o < KK; ++o) {
      float d0 = (p0 < KK) ? pd[0][t][p0] : inf;
      float d1 = (p1 < KK) ? pd[1][t][p1] : inf;
      float d2 = (p2 < KK) ? pd[2][t][p2] : inf;
      float d3 = (p3 < KK) ? pd[3][t][p3] : inf;
      int idx;
      if (d0 <= d1 && d0 <= d2 && d0 <= d3)      { idx = pi[0][t][p0]; ++p0; }
      else if (d1 <= d2 && d1 <= d3)             { idx = pi[1][t][p1]; ++p1; }
      else if (d2 <= d3)                         { idx = pi[2][t][p2]; ++p2; }
      else                                       { idx = pi[3][t][p3]; ++p3; }
      knn[obase + o] = idx;
    }
  }
}

// ---------------- kernel 3: fused per-point transformer ----------------------
// One point per workgroup. 256 threads = 8 waves; wave w owns output cols
// [16w,16w+16) and both 16-row tiles (rows 0..31, 24 valid).
__global__ __launch_bounds__(256) void fused_pt(
    const float* __restrict__ xyz, const float* __restrict__ xbuf,
    const int* __restrict__ knn,
    const __bf16* __restrict__ d1t, const __bf16* __restrict__ d2t,
    const __bf16* __restrict__ g1t, const __bf16* __restrict__ g2t,
    const float* __restrict__ d1b, const float* __restrict__ d2b,
    const float* __restrict__ g1b, const float* __restrict__ g2b,
    const float* __restrict__ fc2w, const float* __restrict__ fc2b,
    float* __restrict__ out_res, float* __restrict__ out_attn) {
  const int p = blockIdx.x;                  // 0 .. BB*NN-1
  const int b = p / NN, n = p % NN;
  const int tid = threadIdx.x;
  const int lane = tid & 31, wv = tid >> 5;
  const int n0 = wv * 16;

  __shared__ int    sidx[KK];
  __shared__ float  gx[KK][3];
  __shared__ float  xc[DM];
  __shared__ __align__(16) __bf16 Abuf[32 * LDA];
  __shared__ float  kfeat[KK][DM];
  __shared__ float  posb[KK][DM];
  __shared__ float  attnb[KK][DM];
  __shared__ float  resb[DM];

  // stage 1: neighbor indices, center features, relative coordinates
  if (tid < KK) sidx[tid] = knn[(size_t)p * KK + tid];
  if (tid < DM) xc[tid] = xbuf[(size_t)p * DM + tid];
  __syncthreads();
  if (tid < KK * 3) {
    int k = tid / 3, c = tid % 3;
    gx[k][c] = xyz[((size_t)b * NN + n) * 3 + c] -
               xyz[((size_t)b * NN + sidx[k]) * 3 + c];
  }
  __syncthreads();

  // stage 2: sincos positional encoding -> Abuf [32][64] bf16 (zero-padded),
  //          gather neighbor features -> kfeat f32
  for (int e = tid; e < 32 * 64; e += 256) {
    int r = e >> 6, c = e & 63;
    float v = 0.0f;
    if (r < KK && c < PE) {
      int cc = c / 20, h = c % 20;
      float omega = __powf(10000.0f, -(float)(h % 10) * 0.1f);
      float t = gx[r][cc] * omega;
      v = (h < 10) ? __sinf(t) : __cosf(t);
    }
    Abuf[r * LDA + c] = (__bf16)v;
  }
  for (int e = tid; e < KK * DM; e += 256) {
    int k = e >> 7, c = e & 127;
    kfeat[k][c] = xbuf[((size_t)b * NN + sidx[k]) * DM + c];
  }
  __syncthreads();

  v8f a0, a1;
  const int col = n0 + (lane & 15);
  const int rb  = (lane >> 4) << 3;          // row offset 0 or 8 within tile

  // GEMM1: h1 = relu(pe @ d1 + d1_b)  (K=64 padded)
  gemm_acc(Abuf, d1t, 64, lane, n0, a0, a1);
  __syncthreads();
  {
    float bias = d1b[col];
#pragma unroll
    for (int r = 0; r < 8; ++r) {
      float h0 = a0[r] + bias; h0 = h0 > 0.0f ? h0 : 0.0f;
      float h1 = a1[r] + bias; h1 = h1 > 0.0f ? h1 : 0.0f;
      Abuf[(rb + r) * LDA + col]      = (__bf16)h0;
      Abuf[(16 + rb + r) * LDA + col] = (__bf16)h1;
    }
  }
  __syncthreads();

  // GEMM2: pos_enc = h1 @ d2 + d2_b ; attn_in = x - k_feat + pos_enc
  gemm_acc(Abuf, d2t, DM, lane, n0, a0, a1);
  __syncthreads();
  {
    float bias = d2b[col], xcv = xc[col];
#pragma unroll
    for (int r = 0; r < 8; ++r) {
      int r0 = rb + r, r1 = 16 + rb + r;
      float p0 = a0[r] + bias, p1 = a1[r] + bias;
      float ai0 = xcv - kfeat[r0][col] + p0;   // r0 < 16 always valid
      posb[r0][col] = p0;
      float ai1 = 0.0f;
      if (r1 < KK) { posb[r1][col] = p1; ai1 = xcv - kfeat[r1][col] + p1; }
      Abuf[r0 * LDA + col] = (__bf16)ai0;
      Abuf[r1 * LDA + col] = (__bf16)ai1;      // pad rows stay zero
    }
  }
  __syncthreads();

  // GEMM3: h2 = relu(attn_in @ g1 + g1_b)
  gemm_acc(Abuf, g1t, DM, lane, n0, a0, a1);
  __syncthreads();
  {
    float bias = g1b[col];
#pragma unroll
    for (int r = 0; r < 8; ++r) {
      float h0 = a0[r] + bias; h0 = h0 > 0.0f ? h0 : 0.0f;
      float h1 = a1[r] + bias; h1 = h1 > 0.0f ? h1 : 0.0f;
      Abuf[(rb + r) * LDA + col]      = (__bf16)h0;
      Abuf[(16 + rb + r) * LDA + col] = (__bf16)h1;
    }
  }
  __syncthreads();

  // GEMM4: logits = (h2 @ g2 + g2_b) / sqrt(D)
  gemm_acc(Abuf, g2t, DM, lane, n0, a0, a1);
  __syncthreads();
  {
    const float sc = 0.08838834764831845f;     // 1/sqrt(128)
    float bias = g2b[col];
#pragma unroll
    for (int r = 0; r < 8; ++r) {
      int r0 = rb + r, r1 = 16 + rb + r;
      attnb[r0][col] = (a0[r] + bias) * sc;
      if (r1 < KK) attnb[r1][col] = (a1[r] + bias) * sc;
    }
  }
  __syncthreads();

  // softmax over K (per feature column) + weighted sum with (v + pos_enc)
  if (tid < DM) {
    int d = tid;
    float m = -__builtin_inff();
#pragma unroll
    for (int k = 0; k < KK; ++k) m = fmaxf(m, attnb[k][d]);
    float s = 0.0f;
#pragma unroll
    for (int k = 0; k < KK; ++k) s += __expf(attnb[k][d] - m);
    float inv = 1.0f / s;
    float r = 0.0f;
    size_t abase = (size_t)p * KK * DM + d;
#pragma unroll
    for (int k = 0; k < KK; ++k) {
      float a = __expf(attnb[k][d] - m) * inv;
      out_attn[abase + (size_t)k * DM] = a;
      r += a * (kfeat[k][d] + posb[k][d]);
    }
    resb[d] = r;
  }
  __syncthreads();

  // final: out = res @ fc2_w + fc2_b + x
  if (tid < DM) {
    int d = tid;
    float o = fc2b[d] + xc[d];
    for (int j = 0; j < DM; ++j) o += resb[j] * fc2w[j * DM + d];
    out_res[(size_t)p * DM + d] = o;
  }
}

// ---------------- host launcher ----------------------------------------------
extern "C" void kernel_launch(void* const* d_in, const int* in_sizes, int n_in,
                              void* d_out, int out_size, void* d_ws,
                              size_t ws_size, hipStream_t stream) {
  (void)in_sizes; (void)n_in; (void)out_size; (void)ws_size;
  const float* features = (const float*)d_in[0];
  const float* xyz      = (const float*)d_in[1];
  const float* fc1_w    = (const float*)d_in[2];
  const float* fc1_b    = (const float*)d_in[3];
  const float* fc2_w    = (const float*)d_in[4];
  const float* fc2_b    = (const float*)d_in[5];
  const float* d1_w     = (const float*)d_in[6];
  const float* d1_b     = (const float*)d_in[7];
  const float* d2_w     = (const float*)d_in[8];
  const float* d2_b     = (const float*)d_in[9];
  const float* g1_w     = (const float*)d_in[10];
  const float* g1_b     = (const float*)d_in[11];
  const float* g2_w     = (const float*)d_in[12];
  const float* g2_b     = (const float*)d_in[13];

  char* ws = (char*)d_ws;
  size_t off = 0;
  auto carve = [&](size_t bytes) {
    void* q = ws + off;
    off = (off + bytes + 255) & ~(size_t)255;
    return q;
  };
  float*  xbuf = (float*)carve((size_t)BB * NN * DM * sizeof(float));
  int*    knn  = (int*)carve((size_t)BB * NN * KK * sizeof(int));
  __bf16* d1t  = (__bf16*)carve((size_t)DM * 64 * sizeof(__bf16));
  __bf16* d2t  = (__bf16*)carve((size_t)DM * DM * sizeof(__bf16));
  __bf16* g1t  = (__bf16*)carve((size_t)DM * DM * sizeof(__bf16));
  __bf16* g2t  = (__bf16*)carve((size_t)DM * DM * sizeof(__bf16));

  float* out_res  = (float*)d_out;
  float* out_attn = out_res + (size_t)BB * NN * DM;

  wconv_kernel<<<64, 256, 0, stream>>>(d1_w, d2_w, g1_w, g2_w,
                                       d1t, d2t, g1t, g2t);
  xproj_kernel<<<BB * NN, 128, 0, stream>>>(features, fc1_w, fc1_b, xbuf);
  knn_kernel<<<(BB * NN) / QPB, 256, 0, stream>>>(xyz, knn);
  fused_pt<<<BB * NN, 256, 0, stream>>>(xyz, xbuf, knn,
                                        d1t, d2t, g1t, g2t,
                                        d1_b, d2_b, g1_b, g2_b,
                                        fc2_w, fc2_b, out_res, out_attn);
}
